// MapLatentEncoder_69252052681248
// MI455X (gfx1250) — matrix operand order repaired and checked by reading, not compiled, not run on hardware
//
#include <hip/hip_runtime.h>
#include <hip/hip_fp16.h>

// MapLatentEncoder, MI455X (gfx1250) version.
// Sizes: bs=64, A=32, T=128, M=64, D=128. N = bs*T*M = 524288 rows.
//
// Algebraic trick: map_feature = codebook[idx] + posemb[m] + ape[bt]  (3 additive
// low-cardinality terms) =>  map_feature @ W0 = C'[idx] + P'[m] + A'[bt] with
// C' = codebook@W0 [128,512], P' = posemb@W0 [64,512], A' = ape@W0 + b0 [8192,512].
// Only the 512->128 matmul remains hot -> v_wmma_f32_16x16x32_f16, with the
// f16 B fragments staged once per workgroup into LDS via
// global_load_async_to_lds_b128 (ASYNCcnt), overlapped with the gather/LN phase.
//
// Workspace usage: ~21.5 MB (posemb 32KB, ape 4MB, Cp 256KB, Pp 128KB,
// Ap 16MB, Bfrag 128KB).

typedef __attribute__((ext_vector_type(16))) _Float16 v16h;
typedef __attribute__((ext_vector_type(8)))  _Float16 v8h;
typedef __attribute__((ext_vector_type(8)))  float    v8f;

#define LN_EPS 1e-5f

__device__ __forceinline__ float wave_sum32(float v) {
#pragma unroll
  for (int m = 16; m >= 1; m >>= 1) v += __shfl_xor(v, m, 32);
  return v;
}

// Block-wide (128 threads = 4 waves) mean/var of one value per thread.
__device__ __forceinline__ void block_stats128(float x, float* red,
                                               float& mean, float& var) {
  const int wv = threadIdx.x >> 5;
  float s = wave_sum32(x);
  float q = wave_sum32(x * x);
  if ((threadIdx.x & 31) == 0) { red[wv] = s; red[4 + wv] = q; }
  __syncthreads();
  float ts = red[0] + red[1] + red[2] + red[3];
  float tq = red[4] + red[5] + red[6] + red[7];
  mean = ts * (1.0f / 128.0f);
  var  = tq * (1.0f / 128.0f) - mean * mean;
  __syncthreads();
}

// ---------------------------------------------------------------------------
// map_pos_mlp: [64,256] @ [256,128] -> LN -> ReLU -> @ [128,128] => posemb[64,128]
__global__ void __launch_bounds__(128) k_mpm(
    const float* __restrict__ pe, const float* __restrict__ w0,
    const float* __restrict__ b0, const float* __restrict__ g0,
    const float* __restrict__ t0, const float* __restrict__ w1,
    const float* __restrict__ b1, float* __restrict__ posemb) {
  __shared__ float sh[256];
  __shared__ float red[8];
  const int j = threadIdx.x, row = blockIdx.x;
  sh[j]       = pe[row * 256 + j];
  sh[j + 128] = pe[row * 256 + 128 + j];
  __syncthreads();
  float a = b0[j];
  for (int k = 0; k < 256; ++k) a += sh[k] * w0[k * 128 + j];
  float m, v;
  block_stats128(a, red, m, v);
  float h = fmaxf(0.0f, (a - m) * rsqrtf(v + LN_EPS) * g0[j] + t0[j]);
  sh[j] = h;
  __syncthreads();
  float o = b1[j];
  for (int k = 0; k < 128; ++k) o += sh[k] * w1[k * 128 + j];
  posemb[row * 128 + j] = o;
}

// ---------------------------------------------------------------------------
// agent_pos_emb: per (b,t): 5 -> 128 LN ReLU -> 128 LN ReLU -> 128 => ape[8192,128]
__global__ void __launch_bounds__(128) k_ape(
    const float* __restrict__ heading, const float* __restrict__ position,
    const float* __restrict__ w0, const float* __restrict__ b0,
    const float* __restrict__ g0, const float* __restrict__ t0,
    const float* __restrict__ w1, const float* __restrict__ b1,
    const float* __restrict__ g1, const float* __restrict__ t1,
    const float* __restrict__ w2, const float* __restrict__ b2,
    float* __restrict__ ape) {
  __shared__ float sh[128];
  __shared__ float red[8];
  const int j = threadIdx.x, bt = blockIdx.x;
  const int b = bt >> 7, t = bt & 127;
  const float hd = heading[b * 4096 + t];          // heading[b,0,t], A*T=4096
  const float p0 = position[b * 8192 + t * 2];     // position[b,0,t,0]
  const float p1 = position[b * 8192 + t * 2 + 1];
  const float f[5] = {p0, p1, hd, __sinf(hd), __cosf(hd)};
  float a = b0[j];
#pragma unroll
  for (int i = 0; i < 5; ++i) a += f[i] * w0[i * 128 + j];
  float m, v;
  block_stats128(a, red, m, v);
  a = fmaxf(0.0f, (a - m) * rsqrtf(v + LN_EPS) * g0[j] + t0[j]);
  sh[j] = a;
  __syncthreads();
  float a1 = b1[j];
  for (int k = 0; k < 128; ++k) a1 += sh[k] * w1[k * 128 + j];
  block_stats128(a1, red, m, v);
  a1 = fmaxf(0.0f, (a1 - m) * rsqrtf(v + LN_EPS) * g1[j] + t1[j]);
  sh[j] = a1;
  __syncthreads();
  float a2 = b2[j];
  for (int k = 0; k < 128; ++k) a2 += sh[k] * w2[k * 128 + j];
  ape[bt * 128 + j] = a2;
}

// ---------------------------------------------------------------------------
// One row of src[128] @ W0[128,512] (+bias). Blocks 0..8191: Ap=ape@W0+b0;
// 8192..8319: Cp=codebook@W0; 8320..8383: Pp=posemb@W0.
__global__ void __launch_bounds__(128) k_rowmm(
    const float* __restrict__ ape, const float* __restrict__ codebook,
    const float* __restrict__ posemb, const float* __restrict__ w0,
    const float* __restrict__ bias0, float* __restrict__ Ap,
    float* __restrict__ Cp, float* __restrict__ Pp) {
  __shared__ float s[128];
  const int j = threadIdx.x, r = blockIdx.x;
  const float* src;
  float* dst;
  bool useBias = false;
  if (r < 8192) {
    src = ape + r * 128; dst = Ap + r * 512; useBias = true;
  } else if (r < 8192 + 128) {
    src = codebook + (r - 8192) * 128; dst = Cp + (r - 8192) * 512;
  } else {
    src = posemb + (r - 8320) * 128; dst = Pp + (r - 8320) * 512;
  }
  s[j] = src[j];
  __syncthreads();
#pragma unroll
  for (int q = 0; q < 4; ++q) {
    const int c = j + q * 128;
    float acc = useBias ? bias0[c] : 0.0f;
    for (int k = 0; k < 128; ++k) acc += s[k] * w0[k * 512 + c];
    dst[c] = acc;
  }
}

// ---------------------------------------------------------------------------
// Repack mla_w1 [512,128] f32 into f16 WMMA B-fragments, stored as two
// 16B-per-lane planes per (nt,kc) so ds_load_b128 reads are bank-optimal:
//   frag[(nt*16+kc)*512 + h*256 + lane*8 + e] =
//       W1[kc*32 + (lane>=16?16:0) + h*8 + e][nt*16 + (lane&15)]
__global__ void __launch_bounds__(256) k_w1frag(const float* __restrict__ w1,
                                                _Float16* __restrict__ frag) {
  const int tid  = blockIdx.x * 256 + threadIdx.x;  // 65536 total
  const int e    = tid & 7;
  const int lane = (tid >> 3) & 31;
  const int h    = (tid >> 8) & 1;
  const int kc   = (tid >> 9) & 15;
  const int nt   = tid >> 13;
  const int n = nt * 16 + (lane & 15);
  const int k = kc * 32 + ((lane & 16) ? 16 : 0) + h * 8 + e;
  frag[tid] = (_Float16)w1[k * 128 + n];
}

// ---------------------------------------------------------------------------
// Main: 8 waves/block, one 16-row tile per wave.
//  - at entry: async-copy all B fragments (128KB) global -> LDS (ASYNCcnt)
//  - phase 1: gather C'[idx]+P'[m]+A'[bt], LN(512)+ReLU -> f16 LDS staging
//  - s_wait_asynccnt 0 + barrier
//  - phase 2: 16 K-chunks x 8 N-tiles of v_wmma_f32_16x16x32_f16, A and B
//    both fed by ds_load_b128.
__global__ void __launch_bounds__(256) k_main(
    const int* __restrict__ map_latent, const float* __restrict__ Cp,
    const float* __restrict__ Pp, const float* __restrict__ Ap,
    const float* __restrict__ g0, const float* __restrict__ t0,
    const _Float16* __restrict__ Bfrag, const float* __restrict__ b1,
    float* __restrict__ out) {
  // 520-half row stride: b128 A-fragment reads hit all 64 LDS banks evenly.
  __shared__ __align__(16) _Float16 hbuf[8][16][520];  // 130 KB
  __shared__ __align__(16) _Float16 Bs[65536];         // 128 KB
  const int wv   = threadIdx.x >> 5;
  const int lane = threadIdx.x & 31;
  const long r0  = ((long)blockIdx.x * 8 + wv) * 16;
  const int col0 = lane * 16;  // lane owns 16 of the 512 hidden channels

  // ---- Kick off async B copy: 8192 x 16B chunks across 256 threads ----
  {
    const uint32_t ldsBase = (uint32_t)(uintptr_t)(&Bs[0]);
    const uint64_t gBase   = (uint64_t)(uintptr_t)Bfrag;
#pragma unroll
    for (int i = 0; i < 32; ++i) {
      const uint32_t c  = (uint32_t)threadIdx.x + (uint32_t)i * 256u;
      const uint32_t la = ldsBase + c * 16u;
      const uint64_t ga = gBase + (uint64_t)c * 16u;
      asm volatile("global_load_async_to_lds_b128 %0, %1, off"
                   :
                   : "v"(la), "v"(ga)
                   : "memory");
    }
  }

  float4 g4[4], t4[4];
#pragma unroll
  for (int q = 0; q < 4; ++q) {
    g4[q] = ((const float4*)(g0 + col0))[q];
    t4[q] = ((const float4*)(t0 + col0))[q];
  }
  const float* gp = (const float*)g4;
  const float* tp = (const float*)t4;

  // ---- Phase 1: build h (16 rows x 512) in LDS as f16 ----
  for (int rr = 0; rr < 16; ++rr) {
    const long r  = r0 + rr;
    const int bt  = (int)(r >> 6);
    const int m   = (int)(r & 63);
    const int idx = map_latent[r];
    const float4* c4 = (const float4*)(Cp + idx * 512 + col0);
    const float4* p4 = (const float4*)(Pp + m * 512 + col0);
    const float4* a4 = (const float4*)(Ap + (long)bt * 512 + col0);
    float x[16];
#pragma unroll
    for (int q = 0; q < 4; ++q) {
      const float4 cv = c4[q], pv = p4[q], av = a4[q];
      x[4 * q + 0] = cv.x + pv.x + av.x;
      x[4 * q + 1] = cv.y + pv.y + av.y;
      x[4 * q + 2] = cv.z + pv.z + av.z;
      x[4 * q + 3] = cv.w + pv.w + av.w;
    }
    float s = 0.0f, q2 = 0.0f;
#pragma unroll
    for (int i = 0; i < 16; ++i) { s += x[i]; q2 += x[i] * x[i]; }
    s  = wave_sum32(s);
    q2 = wave_sum32(q2);
    const float mean = s * (1.0f / 512.0f);
    const float var  = q2 * (1.0f / 512.0f) - mean * mean;
    const float inv  = rsqrtf(var + LN_EPS);
    _Float16 h[16];
#pragma unroll
    for (int i = 0; i < 16; ++i)
      h[i] = (_Float16)fmaxf(0.0f, (x[i] - mean) * inv * gp[i] + tp[i]);
    v8h* dstp = (v8h*)&hbuf[wv][rr][col0];
    dstp[0] = *(v8h*)&h[0];
    dstp[1] = *(v8h*)&h[8];
  }

  // ---- Ensure the async B copy landed (all waves) before reading Bs ----
  asm volatile("s_wait_asynccnt 0x0" ::: "memory");
  __syncthreads();

  // ---- Phase 2: h[16x512] @ W1[512x128] via WMMA, all operands from LDS ----
  v8f acc[8];
  const v8f zero = {};
#pragma unroll
  for (int nt = 0; nt < 8; ++nt) acc[nt] = zero;

  const int arow = lane & 15;        // A-matrix row owned by this lane
  const int koff = (lane >> 4) * 16;  // contiguous K layout: hi lanes start at K=16
#pragma unroll 4
  for (int kc = 0; kc < 16; ++kc) {
    const v8h alo = *(const v8h*)&hbuf[wv][arow][kc * 32 + koff];
    const v8h ahi = *(const v8h*)&hbuf[wv][arow][kc * 32 + 8 + koff];
    v16h a;
#pragma unroll
    for (int i = 0; i < 8; ++i) { a[i] = alo[i]; a[i + 8] = ahi[i]; }
#pragma unroll
    for (int nt = 0; nt < 8; ++nt) {
      const int fbase = (nt * 16 + kc) * 512 + lane * 8;
      const v8h blo = *(const v8h*)&Bs[fbase];
      const v8h bhi = *(const v8h*)&Bs[fbase + 256];
      v16h b;
#pragma unroll
      for (int i = 0; i < 8; ++i) { b[i] = blo[i]; b[i + 8] = bhi[i]; }
      acc[nt] = __builtin_amdgcn_wmma_f32_16x16x32_f16(
          false, a, false, b, (short)0, acc[nt], false, false);
    }
  }

  // ---- Store D + bias (C/D layout: VGPR j -> M=j (+8 for hi lanes), N=lane&15)
  const int rbase = (lane >> 4) * 8;
  const int nlane = lane & 15;
#pragma unroll
  for (int nt = 0; nt < 8; ++nt) {
    const int c   = nt * 16 + nlane;
    const float bb = b1[c];
    float* op = out + (r0 + rbase) * 128 + c;
#pragma unroll
    for (int j = 0; j < 8; ++j) op[(long)j * 128] = acc[nt][j] + bb;
  }
}

// ---------------------------------------------------------------------------
extern "C" void kernel_launch(void* const* d_in, const int* in_sizes, int n_in,
                              void* d_out, int out_size, void* d_ws,
                              size_t ws_size, hipStream_t stream) {
  (void)in_sizes; (void)n_in; (void)out_size; (void)ws_size;
  const int*   map_latent = (const int*)d_in[0];
  const float* heading    = (const float*)d_in[1];
  const float* position   = (const float*)d_in[2];
  const float* codebook   = (const float*)d_in[3];
  const float* mlpe       = (const float*)d_in[4];
  const float* ape_w0 = (const float*)d_in[5],  *ape_b0 = (const float*)d_in[6];
  const float* ape_g0 = (const float*)d_in[7],  *ape_t0 = (const float*)d_in[8];
  const float* ape_w1 = (const float*)d_in[9],  *ape_b1 = (const float*)d_in[10];
  const float* ape_g1 = (const float*)d_in[11], *ape_t1 = (const float*)d_in[12];
  const float* ape_w2 = (const float*)d_in[13], *ape_b2 = (const float*)d_in[14];
  const float* mla_w0 = (const float*)d_in[15], *mla_b0 = (const float*)d_in[16];
  const float* mla_g0 = (const float*)d_in[17], *mla_t0 = (const float*)d_in[18];
  const float* mla_w1 = (const float*)d_in[19], *mla_b1 = (const float*)d_in[20];
  const float* mpm_w0 = (const float*)d_in[21], *mpm_b0 = (const float*)d_in[22];
  const float* mpm_g0 = (const float*)d_in[23], *mpm_t0 = (const float*)d_in[24];
  const float* mpm_w1 = (const float*)d_in[25], *mpm_b1 = (const float*)d_in[26];

  char* ws = (char*)d_ws;
  float* posemb = (float*)ws;  ws += 64 * 128 * 4;           // 32 KB
  float* ape    = (float*)ws;  ws += 8192 * 128 * 4;         // 4 MB
  float* Cp     = (float*)ws;  ws += 128 * 512 * 4;          // 256 KB
  float* Pp     = (float*)ws;  ws += 64 * 512 * 4;           // 128 KB
  float* Ap     = (float*)ws;  ws += (size_t)8192 * 512 * 4; // 16 MB
  _Float16* Bfrag = (_Float16*)ws;                           // 128 KB

  k_mpm<<<64, 128, 0, stream>>>(mlpe, mpm_w0, mpm_b0, mpm_g0, mpm_t0,
                                mpm_w1, mpm_b1, posemb);
  k_ape<<<8192, 128, 0, stream>>>(heading, position, ape_w0, ape_b0, ape_g0,
                                  ape_t0, ape_w1, ape_b1, ape_g1, ape_t1,
                                  ape_w2, ape_b2, ape);
  k_w1frag<<<256, 256, 0, stream>>>(mla_w1, Bfrag);
  k_rowmm<<<8384, 128, 0, stream>>>(ape, codebook, posemb, mla_w0, mla_b0,
                                    Ap, Cp, Pp);
  k_main<<<4096, 256, 0, stream>>>(map_latent, Cp, Pp, Ap, mla_g0, mla_t0,
                                   Bfrag, mla_b1, (float*)d_out);
}
